// AttentionHead_65953517798094
// MI455X (gfx1250) — compile-verified, matrix operand order checked by reading, not run on hardware
//
#include <hip/hip_runtime.h>
#include <hip/hip_bf16.h>

typedef __bf16 bf16_t;
typedef __attribute__((ext_vector_type(16))) bf16_t v16bf;
typedef __attribute__((ext_vector_type(8)))  float  v8f;
typedef __attribute__((ext_vector_type(4)))  float  v4f;
typedef __attribute__((ext_vector_type(4)))  unsigned int v4u;
typedef unsigned short u16;

#define EMBED 768
#define HD 64
#define SEQ 4096
#define BATCH 4
#define ROWS (BATCH*SEQ)          // 16384
#define QSCALE (0.125f * 1.4426950408889634f)  // 1/sqrt(64) * log2(e)

union BV { v16bf v; v4u q[2]; u16 s[16]; };

// hardware f32 -> bf16 convert (v_cvt_*bf16_f32), bit pattern as u16
__device__ __forceinline__ u16 f2bf(float f) {
  bf16_t h = (bf16_t)f;
  return __builtin_bit_cast(u16, h);
}

template<int OFF>
__device__ __forceinline__ float swzf(float v) {
  return __int_as_float(__builtin_amdgcn_ds_swizzle(__float_as_int(v), OFF));
}
// group-of-32 swizzle: offset = and(0x1f) | or<<5 | xor<<10
#define SWZ_X1 0x041F
#define SWZ_X2 0x081F
#define SWZ_X4 0x101F
#define SWZ_X8 0x201F

__device__ __forceinline__ float rmax16(float t) {
  t = fmaxf(t, swzf<SWZ_X1>(t));
  t = fmaxf(t, swzf<SWZ_X2>(t));
  t = fmaxf(t, swzf<SWZ_X4>(t));
  t = fmaxf(t, swzf<SWZ_X8>(t));
  return t;
}
__device__ __forceinline__ float rsum16(float t) {
  t += swzf<SWZ_X1>(t);
  t += swzf<SWZ_X2>(t);
  t += swzf<SWZ_X4>(t);
  t += swzf<SWZ_X8>(t);
  return t;
}

// ---------------- Kernel 0: weights fp32 [768][64] -> bf16 transposed [3][64][768]
__global__ void prep_weights(const float* __restrict__ Wq,
                             const float* __restrict__ Wk,
                             const float* __restrict__ Wv,
                             u16* __restrict__ wt) {
  int idx = blockIdx.x * blockDim.x + threadIdx.x;   // 3*64*768 threads
  int x = idx / (HD * EMBED);
  int r = idx - x * (HD * EMBED);
  int n = r / EMBED;
  int d = r - n * EMBED;
  const float* W = (x == 0) ? Wq : (x == 1) ? Wk : Wv;
  wt[idx] = f2bf(W[d * HD + n]);
}

// ---------------- Kernel 1: QKV projection via WMMA bf16
// Outputs: qb [B*S][64] bf16 (pre-scaled), kbuf [B*S][64] bf16, vt [B][64][S] bf16
__global__ void __launch_bounds__(256) qkv_proj(const float* __restrict__ h,
                         const u16* __restrict__ wt,
                         const float* __restrict__ bq,
                         const float* __restrict__ bk,
                         const float* __restrict__ bv,
                         u16* __restrict__ qb,
                         u16* __restrict__ kbuf,
                         u16* __restrict__ vt) {
  __shared__ u16 tr[8 * 1024];          // 16x64 bf16 transpose buffer per wave
  const int wave = threadIdx.x >> 5;
  const int lane = threadIdx.x & 31;
  const int hl = lane >> 4;             // half-wave select
  const int lr = lane & 15;
  const int tile = blockIdx.x * 8 + wave;     // 0..1023
  const int row0 = tile * 16;                 // global row
  const int b = row0 >> 12;
  const int s0 = row0 & (SEQ - 1);

  v8f aq[4], ak[4], av[4];
  #pragma unroll
  for (int t = 0; t < 4; ++t) { aq[t] = (v8f)0.f; ak[t] = (v8f)0.f; av[t] = (v8f)0.f; }

  const float* arow = h + (size_t)(row0 + lr) * EMBED;

  for (int c = 0; c < 24; ++c) {
    const int k0 = c * 32 + hl * 8;
    v4f a0 = *(const v4f*)(arow + k0);
    v4f a1 = *(const v4f*)(arow + k0 + 4);
    v4f a2 = *(const v4f*)(arow + k0 + 16);
    v4f a3 = *(const v4f*)(arow + k0 + 20);
    BV A;
    #pragma unroll
    for (int i = 0; i < 4; ++i) {
      A.v[i]      = (bf16_t)a0[i];
      A.v[4 + i]  = (bf16_t)a1[i];
      A.v[8 + i]  = (bf16_t)a2[i];
      A.v[12 + i] = (bf16_t)a3[i];
    }
    #pragma unroll
    for (int x = 0; x < 3; ++x) {
      #pragma unroll
      for (int t = 0; t < 4; ++t) {
        const u16* wb = wt + (size_t)(x * HD + t * 16 + lr) * EMBED + c * 32 + hl * 8;
        BV Bm;
        Bm.q[0] = *(const v4u*)(wb);
        Bm.q[1] = *(const v4u*)(wb + 16);
        if (x == 0)
          aq[t] = __builtin_amdgcn_wmma_f32_16x16x32_bf16(false, A.v, false, Bm.v, (short)0, aq[t], false, false);
        else if (x == 1)
          ak[t] = __builtin_amdgcn_wmma_f32_16x16x32_bf16(false, A.v, false, Bm.v, (short)0, ak[t], false, false);
        else
          av[t] = __builtin_amdgcn_wmma_f32_16x16x32_bf16(false, A.v, false, Bm.v, (short)0, av[t], false, false);
      }
    }
  }

  // bias + Q prescale
  #pragma unroll
  for (int t = 0; t < 4; ++t) {
    const int n = t * 16 + lr;
    float vbq = bq[n], vbk = bk[n], vbv = bv[n];
    #pragma unroll
    for (int i = 0; i < 8; ++i) {
      aq[t][i] = (aq[t][i] + vbq) * QSCALE;
      ak[t][i] = ak[t][i] + vbk;
      av[t][i] = av[t][i] + vbv;
    }
  }

  // Vt store: contiguous along seq -> 1 b128 per tile
  #pragma unroll
  for (int t = 0; t < 4; ++t) {
    BV p;
    #pragma unroll
    for (int i = 0; i < 8; ++i) p.v[i] = (bf16_t)av[t][i];
    u16* dst = vt + (size_t)(b * HD + t * 16 + lr) * SEQ + s0 + hl * 8;
    *(v4u*)dst = p.q[0];
  }

  // Q, K: C-layout -> LDS transpose -> coalesced b128 row-major stores
  u16* my = tr + wave * 1024;
  #pragma unroll
  for (int t = 0; t < 4; ++t)
    #pragma unroll
    for (int i = 0; i < 8; ++i)
      my[(hl * 8 + i) * 64 + t * 16 + lr] = f2bf(aq[t][i]);
  #pragma unroll
  for (int j = 0; j < 4; ++j) {
    v4u vv = *(v4u*)(my + (lane + j * 32) * 8);
    int qr = (lane + j * 32) >> 3;
    int qc = (lane & 7) * 8;
    *(v4u*)(qb + (size_t)(row0 + qr) * HD + qc) = vv;
  }
  #pragma unroll
  for (int t = 0; t < 4; ++t)
    #pragma unroll
    for (int i = 0; i < 8; ++i)
      my[(hl * 8 + i) * 64 + t * 16 + lr] = f2bf(ak[t][i]);
  #pragma unroll
  for (int j = 0; j < 4; ++j) {
    v4u vv = *(v4u*)(my + (lane + j * 32) * 8);
    int qr = (lane + j * 32) >> 3;
    int qc = (lane & 7) * 8;
    *(v4u*)(kbuf + (size_t)(row0 + qr) * HD + qc) = vv;
  }
}

// ---------------- Kernel 2: flash attention, 16-query tile per wave, 32-key blocks
__global__ void __launch_bounds__(256) flash_attn(const u16* __restrict__ qb,
                           const u16* __restrict__ kbuf,
                           const u16* __restrict__ vt,
                           float* __restrict__ out) {
  __shared__ u16 psh[8 * 512];           // 16x32 bf16 P tile per wave
  const int wave = threadIdx.x >> 5;
  const int lane = threadIdx.x & 31;
  const int hl = lane >> 4;
  const int lr = lane & 15;
  const int qt = blockIdx.x * 8 + wave;  // 0..1023
  const int b = qt >> 8;
  const int m0 = (qt & 255) * 16;        // local seq row

  // Q A-operands for dims 0-31 and 32-63, kept in registers
  const u16* qrow = qb + ((size_t)(b * SEQ) + m0 + lr) * HD;
  BV A0, A1;
  A0.q[0] = *(const v4u*)(qrow + hl * 8);
  A0.q[1] = *(const v4u*)(qrow + 16 + hl * 8);
  A1.q[0] = *(const v4u*)(qrow + 32 + hl * 8);
  A1.q[1] = *(const v4u*)(qrow + 48 + hl * 8);

  v8f o[4];
  #pragma unroll
  for (int t = 0; t < 4; ++t) o[t] = (v8f)0.f;
  float mrun[8], lsum[8];
  #pragma unroll
  for (int i = 0; i < 8; ++i) { mrun[i] = -3.0e38f; lsum[i] = 0.f; }

  u16* my = psh + wave * 512;
  const int nkb = (m0 + 16 + 31) >> 5;

  for (int kb32 = 0; kb32 < nkb; ++kb32) {
    const int key0 = kb32 * 32;
    // prefetch next key block (K rows + V rows) while we compute this one
    if (kb32 + 1 < nkb) {
      const u16* knext = kbuf + ((size_t)(b * SEQ) + key0 + 32 + lr) * HD;
      __builtin_prefetch(knext, 0, 1);
      __builtin_prefetch(knext + 16 * HD, 0, 1);
      const u16* vnext = vt + ((size_t)(b * HD) + lane) * SEQ + key0 + 32;
      __builtin_prefetch(vnext, 0, 1);
      __builtin_prefetch(vnext + 32 * SEQ, 0, 1);
    }
    // ---- scores: two 16x16 C tiles, K=64 split in two bf16 WMMAs each
    v8f sc[2];
    #pragma unroll
    for (int j = 0; j < 2; ++j) {
      const u16* kr = kbuf + ((size_t)(b * SEQ) + key0 + j * 16 + lr) * HD;
      BV B0, B1;
      B0.q[0] = *(const v4u*)(kr + hl * 8);
      B0.q[1] = *(const v4u*)(kr + 16 + hl * 8);
      B1.q[0] = *(const v4u*)(kr + 32 + hl * 8);
      B1.q[1] = *(const v4u*)(kr + 48 + hl * 8);
      v8f z = (v8f)0.f;
      z = __builtin_amdgcn_wmma_f32_16x16x32_bf16(false, A0.v, false, B0.v, (short)0, z, false, false);
      z = __builtin_amdgcn_wmma_f32_16x16x32_bf16(false, A1.v, false, B1.v, (short)0, z, false, false);
      sc[j] = z;
    }
    // ---- online softmax (rows live across the 16 lanes of each half-wave)
    #pragma unroll
    for (int i = 0; i < 8; ++i) {
      const int rowg = m0 + hl * 8 + i;
      const int c0 = key0 + lr;
      const int c1 = key0 + 16 + lr;
      float x0 = (c0 <= rowg) ? sc[0][i] : -3.0e38f;
      float x1 = (c1 <= rowg) ? sc[1][i] : -3.0e38f;
      float mn = fmaxf(mrun[i], rmax16(fmaxf(x0, x1)));
      float al = __builtin_amdgcn_exp2f(mrun[i] - mn);
      float p0 = __builtin_amdgcn_exp2f(x0 - mn);
      float p1 = __builtin_amdgcn_exp2f(x1 - mn);
      lsum[i] = lsum[i] * al + rsum16(p0 + p1);
      mrun[i] = mn;
      o[0][i] *= al; o[1][i] *= al; o[2][i] *= al; o[3][i] *= al;
      my[(hl * 8 + i) * 32 + lr]      = f2bf(p0);
      my[(hl * 8 + i) * 32 + 16 + lr] = f2bf(p1);
    }
    // ---- reload P in A-operand layout (16 queries x 32 keys)
    BV Pa;
    Pa.q[0] = *(v4u*)(my + lr * 32 + hl * 8);
    Pa.q[1] = *(v4u*)(my + lr * 32 + 16 + hl * 8);
    // ---- O += P @ V  (V transposed layout -> contiguous key-axis loads)
    #pragma unroll
    for (int t = 0; t < 4; ++t) {
      const u16* vr = vt + ((size_t)(b * HD) + t * 16 + lr) * SEQ + key0;
      BV Bm;
      Bm.q[0] = *(const v4u*)(vr + hl * 8);
      Bm.q[1] = *(const v4u*)(vr + 16 + hl * 8);
      o[t] = __builtin_amdgcn_wmma_f32_16x16x32_bf16(false, Pa.v, false, Bm.v, (short)0, o[t], false, false);
    }
  }

  // ---- epilogue: normalize and store fp32
  #pragma unroll
  for (int i = 0; i < 8; ++i) {
    float inv = 1.0f / lsum[i];
    const size_t rbase = ((size_t)(b * SEQ) + m0 + hl * 8 + i) * HD;
    #pragma unroll
    for (int t = 0; t < 4; ++t)
      out[rbase + t * 16 + lr] = o[t][i] * inv;
  }
}

extern "C" void kernel_launch(void* const* d_in, const int* in_sizes, int n_in,
                              void* d_out, int out_size, void* d_ws, size_t ws_size,
                              hipStream_t stream) {
  const float* hidden = (const float*)d_in[0];
  const float* Wq = (const float*)d_in[1];
  const float* bq = (const float*)d_in[2];
  const float* Wk = (const float*)d_in[3];
  const float* bk = (const float*)d_in[4];
  const float* Wv = (const float*)d_in[5];
  const float* bv = (const float*)d_in[6];
  float* out = (float*)d_out;

  // workspace layout (bf16 = u16 elements)
  u16* wt   = (u16*)d_ws;                       // 3*64*768
  u16* qb   = wt + 3 * HD * EMBED;              // 16384*64
  u16* kbuf = qb + (size_t)ROWS * HD;
  u16* vtm  = kbuf + (size_t)ROWS * HD;         // [B][64][S]

  prep_weights<<<(3 * HD * EMBED) / 256, 256, 0, stream>>>(Wq, Wk, Wv, wt);
  qkv_proj<<<ROWS / 16 / 8, 256, 0, stream>>>(hidden, wt, bq, bk, bv, qb, kbuf, vtm);
  flash_attn<<<ROWS / 16 / 8, 256, 0, stream>>>(qb, kbuf, vtm, out);
}
